// Head_37658273251436
// MI455X (gfx1250) — matrix operand order repaired and checked by reading, not compiled, not run on hardware
//
#include <hip/hip_runtime.h>
#include <hip/hip_bf16.h>

// Single-head causal attention, flash-style, via v_wmma_f32_16x16x32_f16,
// with double-buffered GLOBAL_LOAD_ASYNC_TO_LDS staging (ASYNCcnt).
// B=4 T=4096 C=768 H=64.

typedef __attribute__((ext_vector_type(16))) _Float16 v16h;
typedef __attribute__((ext_vector_type(8)))  _Float16 v8h;
typedef __attribute__((ext_vector_type(8)))  float    v8f;
typedef int v4i __attribute__((vector_size(16)));     // matches builtin param type

constexpr int Bz = 4;
constexpr int Tz = 4096;
constexpr int Cz = 768;
constexpr int Hz = 64;

#if __has_builtin(__builtin_amdgcn_global_load_async_to_lds_b128) && \
    __has_builtin(__builtin_amdgcn_s_wait_asynccnt)
#define HAS_ASYNC_LDS 1
#else
#define HAS_ASYNC_LDS 0
#endif

// 16-byte global -> LDS copy: async DMA when available, else direct.
__device__ __forceinline__ void copy16(const void* g, void* l) {
#if HAS_ASYNC_LDS
  __builtin_amdgcn_global_load_async_to_lds_b128(
      (__attribute__((address_space(1))) v4i*)g,
      (__attribute__((address_space(3))) v4i*)l, 0, 0);
#else
  *(v4i*)l = *(const v4i*)g;
#endif
}

template <int N>
__device__ __forceinline__ void wait_async() {
#if HAS_ASYNC_LDS
  __builtin_amdgcn_s_wait_asynccnt(N);
#endif
}

// ---- WMMA fragment gathers (wave32 layouts per CDNA5 ISA 7.12.2) ----------

// A matrix 16x32 f16 from f16 LDS [16 rows][ld halves].
// lane L, elem e: M = L&15, K = (e&7) + 16*(e>>3) + 8*(L>>4)
__device__ __forceinline__ v16h load_a_frag(const _Float16* base, int ld, int lane) {
  const int m  = lane & 15;
  const int kh = (lane >> 4) << 3;
  const _Float16* row = base + m * ld;
  v8h lo = *(const v8h*)(row + kh);
  v8h hi = *(const v8h*)(row + 16 + kh);
  v16h a;
#pragma unroll
  for (int i = 0; i < 8; ++i) { a[i] = lo[i]; a[8 + i] = hi[i]; }
  return a;
}

// Same A gather but from f32 LDS (convert while building the fragment).
__device__ __forceinline__ v16h load_a_frag_f32(const float* base, int ld, int lane) {
  const int m  = lane & 15;
  const int kh = (lane >> 4) << 3;
  const float* row = base + m * ld;
  v8f lo = *(const v8f*)(row + kh);
  v8f hi = *(const v8f*)(row + 16 + kh);
  v16h a;
#pragma unroll
  for (int i = 0; i < 8; ++i) { a[i] = (_Float16)lo[i]; a[8 + i] = (_Float16)hi[i]; }
  return a;
}

// B matrix 32x16 f16: lane L, elem e: N = L&15, K = (e&7) + 16*(e>>3) + 8*(L>>4).
// Source stored "N-major": base[n*ld + k], two 8-half runs per lane.
__device__ __forceinline__ v16h load_b_frag(const _Float16* base, int ld, int lane) {
  const _Float16* row = base + (lane & 15) * ld;
  const int kh = (lane >> 4) << 3;
  v8h lo = *(const v8h*)(row + kh);
  v8h hi = *(const v8h*)(row + 16 + kh);
  v16h b;
#pragma unroll
  for (int i = 0; i < 8; ++i) { b[i] = lo[i]; b[8 + i] = hi[i]; }
  return b;
}

// ---------------------------------------------------------------------------
// Kernel 0: pre-convert Wq/Wk/Wv [C,H] f32 -> wT [mat][h][c] f16 (transposed)
// ---------------------------------------------------------------------------
__global__ __launch_bounds__(256) void wprep_kernel(
    const float* __restrict__ Wq, const float* __restrict__ Wk,
    const float* __restrict__ Wv, _Float16* __restrict__ wT) {
  const int idx = blockIdx.x * 256 + threadIdx.x;
  if (idx >= 3 * Hz * Cz) return;
  const int mat = idx / (Hz * Cz);
  const int rem = idx % (Hz * Cz);
  const int h = rem / Cz, c = rem % Cz;
  const float* W = (mat == 0) ? Wq : (mat == 1) ? Wk : Wv;
  wT[idx] = (_Float16)W[(size_t)c * Hz + h];
}

// ---------------------------------------------------------------------------
// Kernel 1: QKV projection, async double-buffered.
// grid = (B*T/64, 3); block = 128 (4 waves).
// q stored [B*T][H] (scaled by 1/sqrt(H)); k stored [B*T][H]; v stored
// TRANSPOSED [B][H][T] so the attention kernel can async-copy V tiles.
// ---------------------------------------------------------------------------
__device__ __forceinline__ void qkv_issue(const float* x, const _Float16* wtb,
                                          int row0, int c0,
                                          float* sxd, _Float16* swd, int tid) {
  // x tile: 64 rows x 32 f32 (16B chunks; 4 per thread)
#pragma unroll
  for (int j = 0; j < 4; ++j) {
    const int id = tid + 128 * j;                  // 0..511
    const int r = id >> 3, c4 = (id & 7) << 2;
    copy16(x + (size_t)(row0 + r) * Cz + c0 + c4, sxd + r * 32 + c4);
  }
  // wT tile: 64 h-rows x 32 f16 (16B chunks; 2 per thread)
#pragma unroll
  for (int j = 0; j < 2; ++j) {
    const int id = tid + 128 * j;                  // 0..255
    const int h = id >> 2, c8 = (id & 3) << 3;
    copy16(wtb + (size_t)h * Cz + c0 + c8, swd + h * 32 + c8);
  }
}

__global__ __launch_bounds__(128) void qkv_proj_kernel(
    const float* __restrict__ x,
    const _Float16* __restrict__ wT,
    _Float16* __restrict__ qkv16) {
  __shared__ __align__(64) float    sx[2][64 * 32];   // x tile (f32)
  __shared__ __align__(64) _Float16 sw[2][64 * 32];   // W^T tile [h][k]

  const int tid  = threadIdx.x;
  const int lane = tid & 31;
  const int wave = tid >> 5;
  const int mat  = blockIdx.y;
  const _Float16* wtb = wT + (size_t)mat * Hz * Cz;
  const int row0 = blockIdx.x * 64;

  v8f acc[4];
#pragma unroll
  for (int nt = 0; nt < 4; ++nt)
#pragma unroll
    for (int i = 0; i < 8; ++i) acc[nt][i] = 0.f;

  qkv_issue(x, wtb, row0, 0, sx[0], sw[0], tid);

  for (int c0 = 0, it = 0; c0 < Cz; c0 += 32, ++it) {
    const int cur = it & 1;
    if (c0 + 32 < Cz) {
      qkv_issue(x, wtb, row0, c0 + 32, sx[cur ^ 1], sw[cur ^ 1], tid);
      wait_async<6>();                      // this tile's 6 chunks done
    } else {
      wait_async<0>();
    }
    __syncthreads();

    const v16h a = load_a_frag_f32(&sx[cur][wave * 16 * 32], 32, lane);
#pragma unroll
    for (int nt = 0; nt < 4; ++nt) {
      const v16h b = load_b_frag(&sw[cur][nt * 16 * 32], 32, lane);
      acc[nt] = __builtin_amdgcn_wmma_f32_16x16x32_f16(
          false, a, false, b, (short)0, acc[nt], false, false);
    }
    __syncthreads();                        // done reading before overwrite
  }

  const int mbase = (lane >> 4) << 3;
  const int n     = lane & 15;
  _Float16* out   = qkv16 + (size_t)mat * Bz * Tz * Hz;
  if (mat == 2) {
    // V stored transposed: [B][H][T]
#pragma unroll
    for (int nt = 0; nt < 4; ++nt)
#pragma unroll
      for (int r = 0; r < 8; ++r) {
        const int row = row0 + wave * 16 + mbase + r;   // row in [B*T]
        const int bb = row >> 12, tt = row & (Tz - 1);
        out[((size_t)bb * Hz + nt * 16 + n) * Tz + tt] = (_Float16)acc[nt][r];
      }
  } else {
    const float scl = (mat == 0) ? 0.125f : 1.0f;       // fold 1/sqrt(H) into q
#pragma unroll
    for (int nt = 0; nt < 4; ++nt)
#pragma unroll
      for (int r = 0; r < 8; ++r) {
        const int row = row0 + wave * 16 + mbase + r;
        out[(size_t)row * Hz + nt * 16 + n] = (_Float16)(acc[nt][r] * scl);
      }
  }
}

// ---------------------------------------------------------------------------
// Kernel 2: causal flash attention, async double-buffered K/V tiles.
// grid = B * (T/64); block = 128 (4 waves); wave w -> 16 query rows.
// ---------------------------------------------------------------------------
__device__ __forceinline__ void attn_issue(const _Float16* kb, const _Float16* vtb,
                                           int kbase, _Float16* sKd, _Float16* sVd,
                                           int tid) {
  // K tile: contiguous 4KB block (32 keys x 64 halves); 2 chunks per thread
  const _Float16* gk = kb + (size_t)kbase * Hz;
#pragma unroll
  for (int j = 0; j < 2; ++j) {
    const int o = (tid + 128 * j) << 3;            // halves
    copy16(gk + o, sKd + o);
  }
  // V^T tile: 64 h-rows x 32 halves (row stride Tz); 2 chunks per thread
#pragma unroll
  for (int j = 0; j < 2; ++j) {
    const int id = tid + 128 * j;                  // 0..255
    const int h = id >> 2, c8 = (id & 3) << 3;
    copy16(vtb + (size_t)h * Tz + kbase + c8, sVd + h * 32 + c8);
  }
}

__global__ __launch_bounds__(128) void flash_attn_kernel(
    const _Float16* __restrict__ q16,
    const _Float16* __restrict__ k16,
    const _Float16* __restrict__ vT16,
    float* __restrict__ outp) {
  __shared__ __align__(64) _Float16 sQ[64 * 64];          // [q][h]
  __shared__ __align__(64) _Float16 sK[2][32 * 64];       // [key][h]
  __shared__ __align__(64) _Float16 sVT[2][64 * 32];      // [h][key]
  __shared__ __align__(64) _Float16 sP[4][16 * 32];       // per-wave P staging

  const int tid  = threadIdx.x;
  const int lane = tid & 31;
  const int wave = tid >> 5;
  const int b    = blockIdx.x / (Tz / 64);
  const int q0   = (blockIdx.x % (Tz / 64)) * 64;
  const int qw   = q0 + wave * 16;

  const _Float16* qb  = q16  + (size_t)b * Tz * Hz;
  const _Float16* kb  = k16  + (size_t)b * Tz * Hz;
  const _Float16* vtb = vT16 + (size_t)b * Hz * Tz;       // [H][T]

  // Q tile 64x64 f16 -> LDS (plain 16B copies, once)
  for (int i = tid; i < 64 * 8; i += 128) {
    const int r = i >> 3, c8 = (i & 7) << 3;
    *(v4i*)&sQ[r * 64 + c8] = *(const v4i*)(qb + (size_t)(q0 + r) * Hz + c8);
  }

  // prime the async pipeline with key tile 0
  attn_issue(kb, vtb, 0, sK[0], sVT[0], tid);
  __syncthreads();

  const v16h aq0 = load_a_frag(&sQ[wave * 16 * 64 + 0],  64, lane);
  const v16h aq1 = load_a_frag(&sQ[wave * 16 * 64 + 32], 64, lane);

  float mrow[8], lrow[8];
  v8f acc[4];
#pragma unroll
  for (int r = 0; r < 8; ++r) { mrow[r] = -1e30f; lrow[r] = 0.f; }
#pragma unroll
  for (int nt = 0; nt < 4; ++nt)
#pragma unroll
    for (int i = 0; i < 8; ++i) acc[nt][i] = 0.f;

  const int n     = lane & 15;
  const int mbase = (lane >> 4) << 3;
  const int nkt   = q0 / 32 + 2;

  for (int kt = 0; kt < nkt; ++kt) {
    const int kbase = kt * 32;
    const int cur   = kt & 1;
    if (kt + 1 < nkt) {
      attn_issue(kb, vtb, kbase + 32, sK[cur ^ 1], sVT[cur ^ 1], tid);
      wait_async<4>();                     // tile kt's 4 chunks done
    } else {
      wait_async<0>();
    }
    __syncthreads();                       // tile kt visible to all waves

    if (kbase <= qw + 15) {                // wave-uniform
      // S = Q * K^T
      v8f s0, s1;
#pragma unroll
      for (int i = 0; i < 8; ++i) { s0[i] = 0.f; s1[i] = 0.f; }
      {
        const v16h bk00 = load_b_frag(&sK[cur][0 * 16 * 64 + 0],  64, lane);
        const v16h bk01 = load_b_frag(&sK[cur][0 * 16 * 64 + 32], 64, lane);
        const v16h bk10 = load_b_frag(&sK[cur][1 * 16 * 64 + 0],  64, lane);
        const v16h bk11 = load_b_frag(&sK[cur][1 * 16 * 64 + 32], 64, lane);
        s0 = __builtin_amdgcn_wmma_f32_16x16x32_f16(false, aq0, false, bk00, (short)0, s0, false, false);
        s0 = __builtin_amdgcn_wmma_f32_16x16x32_f16(false, aq1, false, bk01, (short)0, s0, false, false);
        s1 = __builtin_amdgcn_wmma_f32_16x16x32_f16(false, aq0, false, bk10, (short)0, s1, false, false);
        s1 = __builtin_amdgcn_wmma_f32_16x16x32_f16(false, aq1, false, bk11, (short)0, s1, false, false);
      }

      // causal mask (scale folded into q)
      if (kbase + 31 > qw) {
#pragma unroll
        for (int r = 0; r < 8; ++r) {
          const int qi = qw + mbase + r;
          if (kbase + n > qi)      s0[r] = -1e30f;
          if (kbase + 16 + n > qi) s1[r] = -1e30f;
        }
      }

      // online softmax (16-lane xor reductions stay inside wave32 halves)
#pragma unroll
      for (int r = 0; r < 8; ++r) {
        float v = fmaxf(s0[r], s1[r]);
        v = fmaxf(v, __shfl_xor(v, 1));
        v = fmaxf(v, __shfl_xor(v, 2));
        v = fmaxf(v, __shfl_xor(v, 4));
        v = fmaxf(v, __shfl_xor(v, 8));
        const float mnew = fmaxf(mrow[r], v);
        const float corr = __expf(mrow[r] - mnew);
        mrow[r] = mnew;
        lrow[r] *= corr;
#pragma unroll
        for (int nt = 0; nt < 4; ++nt) acc[nt][r] *= corr;
        const float p0 = __expf(s0[r] - mnew);
        const float p1 = __expf(s1[r] - mnew);
        float rs = p0 + p1;
        rs += __shfl_xor(rs, 1);
        rs += __shfl_xor(rs, 2);
        rs += __shfl_xor(rs, 4);
        rs += __shfl_xor(rs, 8);
        lrow[r] += rs;
        sP[wave][(mbase + r) * 32 + n]      = (_Float16)p0;
        sP[wave][(mbase + r) * 32 + 16 + n] = (_Float16)p1;
      }

      // O += P * V
      const v16h ap = load_a_frag(&sP[wave][0], 32, lane);
#pragma unroll
      for (int nt = 0; nt < 4; ++nt) {
        const v16h bv = load_b_frag(&sVT[cur][nt * 16 * 32], 32, lane);
        acc[nt] = __builtin_amdgcn_wmma_f32_16x16x32_f16(
            false, ap, false, bv, (short)0, acc[nt], false, false);
      }
    }
    __syncthreads();                       // done reading tile kt
  }

  // epilogue: normalize and store f32 output
#pragma unroll
  for (int r = 0; r < 8; ++r) {
    const float inv = 1.0f / lrow[r];
    const int row = qw + mbase + r;
#pragma unroll
    for (int nt = 0; nt < 4; ++nt)
      outp[((size_t)b * Tz + row) * Hz + nt * 16 + n] = acc[nt][r] * inv;
  }
}

// ---------------------------------------------------------------------------
extern "C" void kernel_launch(void* const* d_in, const int* in_sizes, int n_in,
                              void* d_out, int out_size, void* d_ws, size_t ws_size,
                              hipStream_t stream) {
  const float* x  = (const float*)d_in[0];
  const float* Wq = (const float*)d_in[1];
  const float* Wk = (const float*)d_in[2];
  const float* Wv = (const float*)d_in[3];
  float* out = (float*)d_out;

  _Float16* qkv16 = (_Float16*)d_ws;                // 3 * B*T*H f16 = 6 MB
  const size_t plane = (size_t)Bz * Tz * Hz;
  _Float16* wT = qkv16 + 3 * plane;                 // 3*H*C f16 = 288 KB

  wprep_kernel<<<(3 * Hz * Cz + 255) / 256, 256, 0, stream>>>(Wq, Wk, Wv, wT);

  dim3 g1(Bz * Tz / 64, 3);
  qkv_proj_kernel<<<g1, 128, 0, stream>>>(x, wT, qkv16);

  dim3 g2(Bz * (Tz / 64));
  flash_attn_kernel<<<g2, 128, 0, stream>>>(qkv16, qkv16 + plane,
                                            qkv16 + 2 * plane, out);
}
